// SingleInteractionBlock_1288490189572
// MI455X (gfx1250) — matrix operand order repaired and checked by reading, not compiled, not run on hardware
//
#include <hip/hip_runtime.h>
#include <hip/hip_bf16.h>

// ---------------------------------------------------------------------------
// Fused SingleInteractionBlock for gfx1250 (MI455X, wave32, WMMA).
// Per 16-edge tile (one wave):
//   x[16x28] --wmma--> h[16x28] --64x wmma--> tp_w tiles consumed in-register
//   by the einsum contractions; final 16x16 linears in fp32 VALU; atomic
//   scatter-add by receiver.
// ---------------------------------------------------------------------------

typedef __attribute__((ext_vector_type(16))) _Float16 v16h;
typedef __attribute__((ext_vector_type(8)))  float    v8f;
typedef __attribute__((ext_vector_type(4)))  float    v4f;
typedef __attribute__((ext_vector_type(2)))  float    v2f;
typedef __attribute__((ext_vector_type(8)))  _Float16 v8h;

union F16Frag  { v16h v; v4f q[2]; };
union H16Row   { _Float16 h[32]; v4f q[4]; };

#define EMUL      16      // MUL
#define KPAD      32      // MLP_IN (28) padded to 32
#define WCOLS     1024    // 4*MUL*MUL
#define W1T_BYTES 2048    // 32 cols * 32 k * 2B
#define TILE_E    16

__device__ __forceinline__ void wave_lds_sync() {
  // DS ops are issued in-order per wave; this makes prior DS writes visible
  // to subsequent cross-lane DS reads and stops compiler reordering.
  __builtin_amdgcn_wave_barrier();
  asm volatile("s_wait_dscnt 0" ::: "memory");
  __builtin_amdgcn_wave_barrier();
}

// --- prep: build f16, transposed, K-padded copies of mlp_w1 / mlp_w2 -------
// w1t[n][k] = mlp_w1[k][n]   (32x32, zero padded)
// w2t[n][k] = mlp_w2[k][n]   (1024x32, zero padded in k)
__global__ void prep_weights(const float* __restrict__ w1,
                             const float* __restrict__ w2,
                             _Float16* __restrict__ w1t,
                             _Float16* __restrict__ w2t) {
  int n = blockIdx.x * blockDim.x + threadIdx.x;
  if (n < WCOLS) {
    #pragma unroll
    for (int k = 0; k < KPAD; ++k)
      w2t[n * KPAD + k] = (k < 28) ? (_Float16)w2[k * WCOLS + n] : (_Float16)0.f;
  } else if (n < WCOLS + 32) {
    int m = n - WCOLS;
    #pragma unroll
    for (int k = 0; k < KPAD; ++k)
      w1t[m * KPAD + k] = (k < 28 && m < 28) ? (_Float16)w1[k * 28 + m] : (_Float16)0.f;
  }
}

struct WaveBuf {
  alignas(16) _Float16 xh[TILE_E * KPAD]; // x rows, then reused for h rows (1KB)
  // pkA[(m*16+u)*4] = {s1*s2, s1, dot*inv3, v1x}; overlay after contraction:
  //   outs @0 (e*16+v), outv @256 (e*48+v*3+k)
  alignas(16) float pkA[TILE_E * EMUL * 4];   // 4KB
  // pkB[(m*16+u)*2] = {v1y, v1z}
  alignas(16) float pkB[TILE_E * EMUL * 2];   // 2KB
  float s2[TILE_E];
  float v2[TILE_E * 3];
  int   recv[TILE_E];
};

__launch_bounds__(256)
__global__ void fused_edge_kernel(const float* __restrict__ node_attrs,
                                  const float* __restrict__ node_feats,
                                  const float* __restrict__ edge_attrs,
                                  const float* __restrict__ edge_feats,
                                  const int*   __restrict__ eidx,
                                  const float* __restrict__ lin_ws,
                                  const float* __restrict__ lin_wv,
                                  const _Float16* __restrict__ w1t,
                                  const _Float16* __restrict__ w2t,
                                  float* __restrict__ out,
                                  int E, int n_tiles, int wave_stride) {
  __shared__ float lws[256];
  __shared__ float lwv[256];
  __shared__ WaveBuf wb[8];

  const int tid = threadIdx.x;
  for (int i = tid; i < 256; i += blockDim.x) { lws[i] = lin_ws[i]; lwv[i] = lin_wv[i]; }
  __syncthreads();

  const int lane = tid & 31;
  const int wv   = tid >> 5;
  WaveBuf& W = wb[wv];
  const int hi = lane >> 4;   // which half-wave
  const int lo = lane & 15;

  const float INV_SQRT_IN = 0.18898223650461363f;  // 1/sqrt(28)
  const float SQRT2       = 1.4142135623730951f;
  const float INV32       = 0.17677669529663687f;  // 1/sqrt(32)
  const float INV3        = 0.5773502691896258f;   // 1/sqrt(3)
  const float INVFAN      = 0.25f;                 // 1/sqrt(16)
  const float HSCALE      = SQRT2 * INV_SQRT_IN;   // relu scale folded

  const int gwave = blockIdx.x * (blockDim.x >> 5) + wv;

  for (int t = gwave; t < n_tiles; t += wave_stride) {
    // ---------------- tile setup (lanes 0..15, one edge each) -------------
    if (lane < TILE_E) {
      const int e = t * TILE_E + lane;
      H16Row xr;
      v4f* xdst = reinterpret_cast<v4f*>(&W.xh[lane * KPAD]);
      if (e < E) {
        const int s = eidx[e];
        const int r = eidx[E + e];
        const float* ef   = edge_feats + (size_t)e * 8;
        const float* na_s = node_attrs + (size_t)s * 10;
        const float* na_r = node_attrs + (size_t)r * 10;
        #pragma unroll
        for (int i = 0; i < 8;  ++i) xr.h[i]      = (_Float16)ef[i];
        #pragma unroll
        for (int i = 0; i < 10; ++i) xr.h[8 + i]  = (_Float16)na_s[i];
        #pragma unroll
        for (int i = 0; i < 10; ++i) xr.h[18 + i] = (_Float16)na_r[i];
        #pragma unroll
        for (int i = 28; i < KPAD; ++i) xr.h[i] = (_Float16)0.f;
        #pragma unroll
        for (int i = 0; i < 4; ++i) xdst[i] = xr.q[i];

        const float* ea = edge_attrs + (size_t)e * 4;
        const float s2v = ea[0];
        const float v2x = ea[1], v2y = ea[2], v2z = ea[3];
        W.s2[lane] = s2v;
        W.v2[lane * 3 + 0] = v2x; W.v2[lane * 3 + 1] = v2y; W.v2[lane * 3 + 2] = v2z;
        W.recv[lane] = r;

        const float* nf = node_feats + (size_t)s * 64;
        #pragma unroll
        for (int u = 0; u < EMUL; ++u) {
          const float s1 = nf[u];
          const float a = nf[16 + u * 3 + 0];
          const float b = nf[16 + u * 3 + 1];
          const float c = nf[16 + u * 3 + 2];
          v4f qa; qa.x = s1 * s2v; qa.y = s1;
          qa.z = (a * v2x + b * v2y + c * v2z) * INV3; qa.w = a;
          v2f qb; qb.x = b; qb.y = c;
          *reinterpret_cast<v4f*>(&W.pkA[(lane * 16 + u) * 4]) = qa;
          *reinterpret_cast<v2f*>(&W.pkB[(lane * 16 + u) * 2]) = qb;
        }
      } else {
        const v4f z4 = {};
        const v2f z2 = {};
        #pragma unroll
        for (int i = 0; i < 4; ++i) xdst[i] = z4;
        W.s2[lane] = 0.f;
        W.v2[lane * 3 + 0] = 0.f; W.v2[lane * 3 + 1] = 0.f; W.v2[lane * 3 + 2] = 0.f;
        W.recv[lane] = -1;
        #pragma unroll
        for (int u = 0; u < EMUL; ++u) {
          *reinterpret_cast<v4f*>(&W.pkA[(lane * 16 + u) * 4]) = z4;
          *reinterpret_cast<v2f*>(&W.pkB[(lane * 16 + u) * 2]) = z2;
        }
      }
    }
    wave_lds_sync();

    // per-lane s2 for its 8 D-matrix rows (M = j + 8*hi)
    float s2r[8];
    #pragma unroll
    for (int j = 0; j < 8; ++j) s2r[j] = W.s2[j + 8 * hi];

    // ---------------- layer 1: x @ w1t (two 16-col N-tiles) ---------------
    F16Frag a;
    {
      const _Float16* xrow = &W.xh[lo * KPAD];
      a.q[0] = *reinterpret_cast<const v4f*>(xrow + hi * 16);
      a.q[1] = *reinterpret_cast<const v4f*>(xrow + hi * 16 + 8);
    }
    v8f d0, d1;
    {
      const int kb0 = hi * 16;
      const _Float16* brow0 = w1t + (size_t)lo * KPAD + kb0;
      const _Float16* brow1 = w1t + (size_t)(16 + lo) * KPAD + kb0;
      F16Frag b0, b1;
      b0.q[0] = *reinterpret_cast<const v4f*>(brow0);
      b0.q[1] = *reinterpret_cast<const v4f*>(brow0 + 8);
      b1.q[0] = *reinterpret_cast<const v4f*>(brow1);
      b1.q[1] = *reinterpret_cast<const v4f*>(brow1 + 8);
      v8f z = {};
      d0 = __builtin_amdgcn_wmma_f32_16x16x32_f16(false, a.v, false, b0.v, (short)0, z, false, false);
      d1 = __builtin_amdgcn_wmma_f32_16x16x32_f16(false, a.v, false, b1.v, (short)0, z, false, false);
    }
    // h = sqrt(2)/sqrt(28) * max(d, 0), stored f16 over x
    #pragma unroll
    for (int j = 0; j < 8; ++j) {
      const int m = j + 8 * hi;
      const float h0 = HSCALE * fmaxf(d0[j], 0.f);
      const float h1 = HSCALE * fmaxf(d1[j], 0.f);
      W.xh[m * KPAD + lo]      = (_Float16)h0;
      W.xh[m * KPAD + 16 + lo] = (_Float16)h1;
    }
    wave_lds_sync();
    {
      const _Float16* hrow = &W.xh[lo * KPAD];
      a.q[0] = *reinterpret_cast<const v4f*>(hrow + hi * 16);
      a.q[1] = *reinterpret_cast<const v4f*>(hrow + hi * 16 + 8);
    }

    // ------- layer 2 (64 WMMAs) fused with the einsum contractions --------
    float accS[8]  = {0.f,0.f,0.f,0.f,0.f,0.f,0.f,0.f};
    float accSV[8] = {0.f,0.f,0.f,0.f,0.f,0.f,0.f,0.f};
    float accV0[8] = {0.f,0.f,0.f,0.f,0.f,0.f,0.f,0.f};
    float accV1[8] = {0.f,0.f,0.f,0.f,0.f,0.f,0.f,0.f};
    float accV2[8] = {0.f,0.f,0.f,0.f,0.f,0.f,0.f,0.f};
    const int kb = hi * 16;

    for (int u = 0; u < EMUL; ++u) {
      float css[8], cs1[8], cdt[8], cvx[8], cvy[8], cvz[8];
      #pragma unroll
      for (int j = 0; j < 8; ++j) {
        const int m = j + 8 * hi;
        const v4f qa = *reinterpret_cast<const v4f*>(&W.pkA[(m * 16 + u) * 4]);
        const v2f qb = *reinterpret_cast<const v2f*>(&W.pkB[(m * 16 + u) * 2]);
        css[j] = qa.x; cs1[j] = qa.y; cdt[j] = qa.z; cvx[j] = qa.w;
        cvy[j] = qb.x; cvz[j] = qb.y;
      }
      #pragma unroll
      for (int c = 0; c < 4; ++c) {
        const int n0 = c * 256 + u * 16;        // tp_w column block (c,u)
        const _Float16* brow = w2t + (size_t)(n0 + lo) * KPAD + kb;
        F16Frag b;
        b.q[0] = *reinterpret_cast<const v4f*>(brow);
        b.q[1] = *reinterpret_cast<const v4f*>(brow + 8);
        v8f z = {};
        v8f d = __builtin_amdgcn_wmma_f32_16x16x32_f16(false, a.v, false, b.v, (short)0, z, false, false);
        #pragma unroll
        for (int j = 0; j < 8; ++j) {
          if      (c == 0) accS[j]  += d[j] * css[j];   // w_sss (s1*s2)
          else if (c == 1) accSV[j] += d[j] * cs1[j];   // w_svv (s1)
          else if (c == 2) {                            // w_vsv (v1)
            accV0[j] += d[j] * cvx[j];
            accV1[j] += d[j] * cvy[j];
            accV2[j] += d[j] * cvz[j];
          } else           accS[j]  += d[j] * cdt[j];   // w_vvs (dot*inv3)
        }
      }
    }

    // ---------------- finalize into LDS overlay ---------------------------
    wave_lds_sync();
    const float SC = INV32 * INV_SQRT_IN;   // tp_w scale folded in
    #pragma unroll
    for (int j = 0; j < 8; ++j) {
      const int m = j + 8 * hi;
      W.pkA[m * 16 + lo] = SC * accS[j];                          // out_s
      const float sv  = SC * accSV[j];
      const float ss  = SC * s2r[j];
      const float vx = W.v2[m * 3 + 0];
      const float vy = W.v2[m * 3 + 1];
      const float vz = W.v2[m * 3 + 2];
      W.pkA[256 + m * 48 + lo * 3 + 0] = fmaf(sv, vx, ss * accV0[j]);
      W.pkA[256 + m * 48 + lo * 3 + 1] = fmaf(sv, vy, ss * accV1[j]);
      W.pkA[256 + m * 48 + lo * 3 + 2] = fmaf(sv, vz, ss * accV2[j]);
    }
    wave_lds_sync();

    // ---------------- final 16x16 linears + scatter-add --------------------
    for (int idx = lane; idx < TILE_E * 64; idx += 32) {
      const int e = idx >> 6;
      const int c = idx & 63;
      const int rr = W.recv[e];
      if (rr >= 0) {
        float y = 0.f;
        if (c < 16) {
          #pragma unroll
          for (int v = 0; v < 16; ++v) y += W.pkA[e * 16 + v] * lws[v * 16 + c];
        } else {
          const int cc   = c - 16;
          const int wcol = cc / 3;
          const int k    = cc - wcol * 3;
          #pragma unroll
          for (int v = 0; v < 16; ++v)
            y += W.pkA[256 + e * 48 + v * 3 + k] * lwv[v * 16 + wcol];
        }
        atomicAdd(out + (size_t)rr * 64 + c, y * INVFAN);
      }
    }
    wave_lds_sync();   // tile buffers reused next iteration
  }
}

extern "C" void kernel_launch(void* const* d_in, const int* in_sizes, int n_in,
                              void* d_out, int out_size, void* d_ws, size_t ws_size,
                              hipStream_t stream) {
  const float* node_attrs = (const float*)d_in[0];
  const float* node_feats = (const float*)d_in[1];
  const float* edge_attrs = (const float*)d_in[2];
  const float* edge_feats = (const float*)d_in[3];
  const int*   eidx       = (const int*)  d_in[4];
  const float* mlp_w1     = (const float*)d_in[5];
  const float* mlp_w2     = (const float*)d_in[6];
  const float* lin_ws     = (const float*)d_in[7];
  const float* lin_wv     = (const float*)d_in[8];
  float* out = (float*)d_out;

  const int E = in_sizes[3] / 8;            // edge_feats is (E, 8)
  _Float16* w1t = (_Float16*)d_ws;
  _Float16* w2t = (_Float16*)((char*)d_ws + W1T_BYTES);

  hipMemsetAsync(d_out, 0, (size_t)out_size * sizeof(float), stream);

  prep_weights<<<dim3((WCOLS + 32 + 255) / 256), dim3(256), 0, stream>>>(
      mlp_w1, mlp_w2, w1t, w2t);

  const int n_tiles = (E + TILE_E - 1) / TILE_E;
  const int blocks  = 512;                  // 8 waves per block
  const int wave_stride = blocks * 8;
  fused_edge_kernel<<<dim3(blocks), dim3(256), 0, stream>>>(
      node_attrs, node_feats, edge_attrs, edge_feats, eidx,
      lin_ws, lin_wv, w1t, w2t, out, E, n_tiles, wave_stride);
}
